// Graphormer_43413529428777
// MI455X (gfx1250) — compile-verified, matrix operand order checked by reading, not run on hardware
//
#include <hip/hip_runtime.h>
#include <math.h>

typedef __attribute__((ext_vector_type(16))) __bf16 v16bf;
typedef __attribute__((ext_vector_type(8)))  float  v8f;
typedef __attribute__((ext_vector_type(8)))  unsigned short us8;
typedef unsigned short bf16u;

constexpr int N_NODES = 2048;
constexpr int E_EDGES = 32768;
constexpr int IN_ND   = 128;
constexpr int ND      = 256;
constexpr int IN_ED   = 64;
constexpr int ED      = 128;
constexpr int OUT_D   = 256;
constexpr int NH      = 8;
constexpr int FF      = 1024;
constexpr int NL      = 3;
constexpr int MAX_DEG = 64;
constexpr int LP      = 5;
constexpr int DK      = ND;   // per-head dim == ND in this model

__device__ inline unsigned bfbits(float f) {
  union { float f; unsigned u; } x; x.f = f;
  return (x.u + 0x7FFFu + ((x.u >> 16) & 1u)) >> 16;   // RNE f32->bf16
}
__device__ inline unsigned pack2bf(float lo, float hi) {
  return bfbits(lo) | (bfbits(hi) << 16);
}

// Staging helpers: return a packed bf16 pair (2 consecutive K elements).
// f32 source: load float2 + convert.  bf16 source: raw 32-bit copy (no VALU).
__device__ inline unsigned loadPairA(const float* A, long long idx) {
  const float2 f = *(const float2*)(A + idx);
  return pack2bf(f.x, f.y);
}
__device__ inline unsigned loadPairA(const bf16u* A, long long idx) {
  return *(const unsigned*)(A + idx);
}
__device__ inline unsigned loadPairB(const float* B, long long idx, int ldb) {
  return pack2bf(B[idx], B[idx + ldb]);
}
__device__ inline unsigned loadPairB(const bf16u* B, long long idx, int ldb) {
  return (unsigned)B[idx] | ((unsigned)B[idx + ldb] << 16);
}

union Frag { us8 h[2]; v16bf v; };

// ---------------------------------------------------------------------------
// Tiled bf16 WMMA GEMM:  C[M,Ncols] = A[M,K] * B[K,Ncols]   (TA/TB: f32|bf16)
// 256 threads = 8 waves; block tile 64x128, TK=64.
// Wave grid 2x4 super-tiles of 32x32; each wave holds 2x2 16x16 accumulators
// -> 4 fragment loads feed 4 WMMAs (8 WMMAs per K-step).
// A staged [m][k], B staged TRANSPOSED [n][k]: all fragment reads are
// contiguous 16-byte LDS vector loads.
// Requires M%64==0, Ncols%128==0, K%64==0 (true for every GEMM here).
// ---------------------------------------------------------------------------
template <typename TA, typename TB, typename Epi>
__global__ void __launch_bounds__(256)
gemm_wmma(const TA* __restrict__ A, int lda, long long strideA,
          const TB* __restrict__ B, int ldb, long long strideB,
          int M, int Ncols, int K, Epi epi)
{
  __shared__ unsigned short As[64][72];    // 72-half pitch: 144B, 16B-aligned rows
  __shared__ unsigned short Bt[128][72];

  const int z = blockIdx.z;
  A += (long long)z * strideA;
  B += (long long)z * strideB;

  const int tileM = blockIdx.y * 64;
  const int tileN = blockIdx.x * 128;
  const int tid   = threadIdx.x;
  const int wid   = tid >> 5;
  const int lane  = tid & 31;
  const int smr   = (wid >> 2) << 5;   // super-tile row: 0 or 32
  const int snc   = (wid & 3)  << 5;   // super-tile col: 0,32,64,96
  const int lrow  = lane & 15;
  const int khi   = (lane >> 4) << 3;  // 0 or 8 (ISA 16-bit operand lane split)

  v8f acc[2][2];
#pragma unroll
  for (int mi = 0; mi < 2; ++mi)
#pragma unroll
    for (int ni = 0; ni < 2; ++ni)
      acc[mi][ni] = (v8f){0.f, 0.f, 0.f, 0.f, 0.f, 0.f, 0.f, 0.f};

  for (int k0 = 0; k0 < K; k0 += 64) {
    // Stage A: 64 rows x 32 pairs = 2048 pairs, 8 per thread
#pragma unroll
    for (int i = 0; i < 8; ++i) {
      int p  = tid + i * 256;          // 0..2047
      int r  = p >> 5;                 // row 0..63
      int pc = p & 31;                 // pair col 0..31
      *(unsigned*)&As[r][2 * pc] =
          loadPairA(A, (long long)(tileM + r) * lda + k0 + 2 * pc);
    }
    // Stage B transposed: 128 cols x 32 k-pairs = 4096 pairs, 16 per thread.
    // Loads hit two consecutive B rows, coalesced along n; one b32 store each.
#pragma unroll
    for (int i = 0; i < 16; ++i) {
      int p  = tid + i * 256;          // 0..4095
      int n  = p & 127;                // col 0..127
      int kp = p >> 7;                 // k-pair 0..31
      *(unsigned*)&Bt[n][2 * kp] =
          loadPairB(B, (long long)(k0 + 2 * kp) * ldb + tileN + n, ldb);
    }
    __syncthreads();

#pragma unroll
    for (int t = 0; t < 2; ++t) {
      const int kb = 32 * t;
      Frag fa[2], fb[2];
#pragma unroll
      for (int i = 0; i < 2; ++i) {
        fa[i].h[0] = *(const us8*)&As[smr + 16 * i + lrow][kb + khi];
        fa[i].h[1] = *(const us8*)&As[smr + 16 * i + lrow][kb + 16 + khi];
        fb[i].h[0] = *(const us8*)&Bt[snc + 16 * i + lrow][kb + khi];
        fb[i].h[1] = *(const us8*)&Bt[snc + 16 * i + lrow][kb + 16 + khi];
      }
#pragma unroll
      for (int mi = 0; mi < 2; ++mi)
#pragma unroll
        for (int ni = 0; ni < 2; ++ni)
          acc[mi][ni] = __builtin_amdgcn_wmma_f32_16x16x32_bf16(
              false, fa[mi].v, false, fb[ni].v, (short)0, acc[mi][ni], false, false);
    }
    __syncthreads();
  }

  // D layout: VGPR v -> M = v (lanes 0-15) / v+8 (lanes 16-31), N = lane&15
#pragma unroll
  for (int mi = 0; mi < 2; ++mi) {
#pragma unroll
    for (int ni = 0; ni < 2; ++ni) {
      const int n     = tileN + snc + 16 * ni + lrow;
      const int mbase = tileM + smr + 16 * mi + ((lane >> 4) << 3);
#pragma unroll
      for (int v = 0; v < 8; ++v)
        epi(z, mbase + v, n, acc[mi][ni][v]);
    }
  }
}

// ------------------------------- epilogues ---------------------------------
struct EpiBias {           // f32 out[z][m][n] = acc + bias[z][n]
  float* out; long long strideO; int ldc;
  const float* bias; int strideBias;
  __device__ void operator()(int z, int m, int n, float acc) const {
    float v = acc;
    if (bias) v += bias[(long long)z * strideBias + n];
    out[(long long)z * strideO + (long long)m * ldc + n] = v;
  }
};
struct EpiBiasBf {         // bf16 out[z][m][n] = acc + bias[z][n]
  bf16u* out; long long strideO; int ldc;
  const float* bias; int strideBias;
  __device__ void operator()(int z, int m, int n, float acc) const {
    out[(long long)z * strideO + (long long)m * ldc + n] =
        (bf16u)bfbits(acc + bias[(long long)z * strideBias + n]);
  }
};
struct EpiBiasTBf {        // bf16 transposed store: out[z][n][m]  (for K^T)
  bf16u* out; long long strideO; int ldt;
  const float* bias; int strideBias;
  __device__ void operator()(int z, int m, int n, float acc) const {
    out[(long long)z * strideO + (long long)n * ldt + m] =
        (bf16u)bfbits(acc + bias[(long long)z * strideBias + n]);
  }
};
struct EpiBiasRes {        // f32 out[m][n] = acc + bias[n] + res[m][n]
  float* out; int ldc; const float* bias; const float* res; int ldr;
  __device__ void operator()(int, int m, int n, float acc) const {
    out[(long long)m * ldc + n] = acc + bias[n] + res[(long long)m * ldr + n];
  }
};
struct EpiDeg {            // input proj + centrality embeddings (f32)
  float* out; int ldc; const float* bias;
  const float* z_in; const float* z_out;
  const int* din; const int* dout; int maxdeg; int nd;
  __device__ void operator()(int, int m, int n, float acc) const {
    int a = din[m];  if (a > maxdeg - 1) a = maxdeg - 1;
    int b = dout[m]; if (b > maxdeg - 1) b = maxdeg - 1;
    out[(long long)m * ldc + n] =
        acc + bias[n] + z_in[(long long)a * nd + n] + z_out[(long long)b * nd + n];
  }
};
struct EpiGeluBf {         // exact GELU (erf) -> bf16
  bf16u* out; int ldc; const float* bias;
  __device__ void operator()(int, int m, int n, float acc) const {
    float x = acc + bias[n];
    out[(long long)m * ldc + n] =
        (bf16u)bfbits(0.5f * x * (1.f + erff(x * 0.70710678118654752f)));
  }
};
struct EpiScore {          // fused spatial bias + shortest-path edge bias (f32)
  float* out; int Nn; float scale;
  const int* node_dist; const int* edge_paths;
  const float* dots; const float* b_spatial; int Lmax; int Ecnt;
  __device__ void operator()(int z, int m, int n, float acc) const {
    float v = acc * scale;
    int dist = node_dist[(long long)m * Nn + n];
    if (dist > 0) {
      int ds = dist > Lmax ? Lmax : dist;
      v += b_spatial[ds - 1];
      int npe = dist - 1; if (npe > Lmax) npe = Lmax;
      if (npe > 0) {
        const int* p = edge_paths + ((long long)m * Nn + n) * Lmax;
        float cs = 0.f;
        for (int s = 0; s < npe; ++s)
          cs += dots[((long long)z * Lmax + s) * Ecnt + p[s]];
        v += cs / (float)npe;
      }
    }
    out[((long long)z * Nn + m) * (long long)Nn + n] = v;
  }
};
struct EpiOBf {            // bf16 scatter heads: o[m][z*dk + n]
  bf16u* out; int ldo; int dk;
  __device__ void operator()(int z, int m, int n, float acc) const {
    out[(long long)m * ldo + z * dk + n] = (bf16u)bfbits(acc);
  }
};

// ----------------------------- small kernels -------------------------------
__global__ void zero_i32(int* p, int n) {
  int i = blockIdx.x * blockDim.x + threadIdx.x;
  if (i < n) p[i] = 0;
}
__global__ void deg_kernel(const int* __restrict__ ei, int* din, int* dout, int e) {
  int i = blockIdx.x * blockDim.x + threadIdx.x;
  if (i < e) {
    atomicAdd(&dout[ei[i]], 1);      // row 0: src -> out-degree
    atomicAdd(&din[ei[e + i]], 1);   // row 1: dst -> in-degree
  }
}
__global__ void __launch_bounds__(256)
ln_rows(const float* __restrict__ x, bf16u* __restrict__ y,
        const float* __restrict__ w, const float* __restrict__ b, int nd) {
  __shared__ float red[256];
  const float* row = x + (long long)blockIdx.x * nd;
  int t = threadIdx.x;
  float v = (t < nd) ? row[t] : 0.f;
  red[t] = v; __syncthreads();
  for (int s = 128; s > 0; s >>= 1) { if (t < s) red[t] += red[t + s]; __syncthreads(); }
  float mean = red[0] / nd; __syncthreads();
  float d = (t < nd) ? (v - mean) : 0.f;
  red[t] = d * d; __syncthreads();
  for (int s = 128; s > 0; s >>= 1) { if (t < s) red[t] += red[t + s]; __syncthreads(); }
  float rs = rsqrtf(red[0] / nd + 1e-5f);
  if (t < nd) y[(long long)blockIdx.x * nd + t] = (bf16u)bfbits(d * rs * w[t] + b[t]);
}
__global__ void __launch_bounds__(256)
dots_kernel(const float* __restrict__ e, const float* __restrict__ evec,
            float* __restrict__ dots, int ecnt, int ed) {
  __shared__ float vsh[ED];
  if (threadIdx.x < ed) vsh[threadIdx.x] = evec[(long long)blockIdx.y * ed + threadIdx.x];
  __syncthreads();
  int idx = blockIdx.x * blockDim.x + threadIdx.x;
  if (idx < ecnt) {
    const float* er = e + (long long)idx * ed;
    float s = 0.f;
    for (int d = 0; d < ed; ++d) s += er[d] * vsh[d];
    dots[(long long)blockIdx.y * ecnt + idx] = s;
  }
}
// Softmax over each row of a [rows, ncols] f32 tensor, emitting bf16.
// exp is recomputed in the write pass instead of stored: trades cheap
// transcendental VALU for a full f32 round-trip of the 134MB tensor.
__global__ void __launch_bounds__(256)
softmax_rows_bf(const float* __restrict__ a, bf16u* __restrict__ o, int ncols) {
  __shared__ float red[256];
  const float* row = a + (long long)blockIdx.x * ncols;
  bf16u* orow = o + (long long)blockIdx.x * ncols;
  int t = threadIdx.x;
  float mx = -INFINITY;
  for (int j = t; j < ncols; j += 256) mx = fmaxf(mx, row[j]);
  red[t] = mx; __syncthreads();
  for (int s = 128; s > 0; s >>= 1) { if (t < s) red[t] = fmaxf(red[t], red[t + s]); __syncthreads(); }
  mx = red[0]; __syncthreads();
  float sum = 0.f;
  for (int j = t; j < ncols; j += 256) sum += expf(row[j] - mx);
  red[t] = sum; __syncthreads();
  for (int s = 128; s > 0; s >>= 1) { if (t < s) red[t] += red[t + s]; __syncthreads(); }
  float inv = 1.f / red[0];
  for (int j = t; j < ncols; j += 256)
    orow[j] = (bf16u)bfbits(expf(row[j] - mx) * inv);
}

// ------------------------------- launcher ----------------------------------
extern "C" void kernel_launch(void* const* d_in, const int* in_sizes, int n_in,
                              void* d_out, int out_size, void* d_ws, size_t ws_size,
                              hipStream_t stream)
{
  (void)in_sizes; (void)n_in; (void)out_size; (void)ws_size;
  const float* x          = (const float*)d_in[0];
  const int*   edge_index = (const int*)  d_in[1];
  const float* edge_attr  = (const float*)d_in[2];
  const int*   node_dist  = (const int*)  d_in[3];
  const int*   edge_paths = (const int*)  d_in[4];
  const float* Wn_in = (const float*)d_in[5];
  const float* bn_in = (const float*)d_in[6];
  const float* We_in = (const float*)d_in[7];
  const float* be_in = (const float*)d_in[8];
  const float* z_in  = (const float*)d_in[9];
  const float* z_out = (const float*)d_in[10];
  const float* b_spatial = (const float*)d_in[11];
  const float* ln1_w = (const float*)d_in[12];
  const float* ln1_b = (const float*)d_in[13];
  const float* ln2_w = (const float*)d_in[14];
  const float* ln2_b = (const float*)d_in[15];
  const float* Wq = (const float*)d_in[16];
  const float* bq = (const float*)d_in[17];
  const float* Wk = (const float*)d_in[18];
  const float* bk = (const float*)d_in[19];
  const float* Wv = (const float*)d_in[20];
  const float* bv = (const float*)d_in[21];
  const float* edge_vec = (const float*)d_in[22];
  const float* Wo = (const float*)d_in[23];
  const float* bo = (const float*)d_in[24];
  const float* Wff1 = (const float*)d_in[25];
  const float* bff1 = (const float*)d_in[26];
  const float* Wff2 = (const float*)d_in[27];
  const float* bff2 = (const float*)d_in[28];
  const float* Wout = (const float*)d_in[29];
  const float* bout = (const float*)d_in[30];

  char* ws = (char*)d_ws;
  size_t off = 0;
  auto take = [&](size_t bytes) -> char* {
    char* p = ws + off;
    off += (bytes + 255) & ~(size_t)255;
    return p;
  };
  int*   deg_in  = (int*)  take(sizeof(int)   * N_NODES);
  int*   deg_out = (int*)  take(sizeof(int)   * N_NODES);
  float* hbuf    = (float*)take(sizeof(float) * (size_t)N_NODES * ND);
  bf16u* xnb     = (bf16u*)take(sizeof(bf16u) * (size_t)N_NODES * ND);
  float* ebuf    = (float*)take(sizeof(float) * (size_t)E_EDGES * ED);
  bf16u* qbuf    = (bf16u*)take(sizeof(bf16u) * (size_t)NH * N_NODES * DK);
  bf16u* kTb     = (bf16u*)take(sizeof(bf16u) * (size_t)NH * DK * N_NODES);
  bf16u* vbuf    = (bf16u*)take(sizeof(bf16u) * (size_t)NH * N_NODES * DK);
  float* dotsb   = (float*)take(sizeof(float) * (size_t)NH * LP * E_EDGES);
  float* attn    = (float*)take(sizeof(float) * (size_t)NH * N_NODES * N_NODES);
  bf16u* attnbf  = (bf16u*)take(sizeof(bf16u) * (size_t)NH * N_NODES * N_NODES);
  bf16u* obuf    = (bf16u*)take(sizeof(bf16u) * (size_t)N_NODES * NH * DK);
  bf16u* ffb     = (bf16u*)take(sizeof(bf16u) * (size_t)N_NODES * FF);

  dim3 blk(256);

  zero_i32<<<dim3((N_NODES + 255) / 256), blk, 0, stream>>>(deg_in,  N_NODES);
  zero_i32<<<dim3((N_NODES + 255) / 256), blk, 0, stream>>>(deg_out, N_NODES);
  deg_kernel<<<dim3((E_EDGES + 255) / 256), blk, 0, stream>>>(edge_index, deg_in, deg_out, E_EDGES);

  { // h = x @ Wn_in + bn_in + z_in[deg_in] + z_out[deg_out]
    EpiDeg epi{hbuf, ND, bn_in, z_in, z_out, deg_in, deg_out, MAX_DEG, ND};
    gemm_wmma<<<dim3(ND / 128, N_NODES / 64, 1), blk, 0, stream>>>(
        x, IN_ND, 0LL, Wn_in, ND, 0LL, N_NODES, ND, IN_ND, epi);
  }
  { // e = edge_attr @ We_in + be_in
    EpiBias epi{ebuf, 0LL, ED, be_in, 0};
    gemm_wmma<<<dim3(ED / 128, E_EDGES / 64, 1), blk, 0, stream>>>(
        edge_attr, IN_ED, 0LL, We_in, ED, 0LL, E_EDGES, ED, IN_ED, epi);
  }

  const float scale = 1.0f / sqrtf((float)DK);

  for (int l = 0; l < NL; ++l) {
    ln_rows<<<dim3(N_NODES), blk, 0, stream>>>(hbuf, xnb, ln1_w + (size_t)l * ND, ln1_b + (size_t)l * ND, ND);

    { // Q[h] = xn @ Wq[l,h] + bq   -> bf16
      EpiBiasBf epi{qbuf, (long long)N_NODES * DK, DK, bq + (size_t)l * NH * DK, DK};
      gemm_wmma<<<dim3(DK / 128, N_NODES / 64, NH), blk, 0, stream>>>(
          xnb, ND, 0LL, Wq + (size_t)l * NH * ND * ND, ND, (long long)ND * ND,
          N_NODES, DK, ND, epi);
    }
    { // K^T[h] stored bf16 [dk][N] for row-major score GEMM B operand
      EpiBiasTBf epi{kTb, (long long)DK * N_NODES, N_NODES, bk + (size_t)l * NH * DK, DK};
      gemm_wmma<<<dim3(DK / 128, N_NODES / 64, NH), blk, 0, stream>>>(
          xnb, ND, 0LL, Wk + (size_t)l * NH * ND * ND, ND, (long long)ND * ND,
          N_NODES, DK, ND, epi);
    }
    { // V[h] -> bf16
      EpiBiasBf epi{vbuf, (long long)N_NODES * DK, DK, bv + (size_t)l * NH * DK, DK};
      gemm_wmma<<<dim3(DK / 128, N_NODES / 64, NH), blk, 0, stream>>>(
          xnb, ND, 0LL, Wv + (size_t)l * NH * ND * ND, ND, (long long)ND * ND,
          N_NODES, DK, ND, epi);
    }

    // dots[h,s,e] = edge_vec[l,h,s,:] . e[e,:]   (5 MB table, L2-resident)
    dots_kernel<<<dim3((E_EDGES + 255) / 256, NH * LP), blk, 0, stream>>>(
        ebuf, edge_vec + (size_t)l * NH * LP * ED, dotsb, E_EDGES, ED);

    { // scores = Q K^T * scale + spatial bias + path-edge bias (fused epilogue)
      EpiScore epi{attn, N_NODES, scale, node_dist, edge_paths, dotsb, b_spatial, LP, E_EDGES};
      gemm_wmma<<<dim3(N_NODES / 128, N_NODES / 64, NH), blk, 0, stream>>>(
          qbuf, DK, (long long)N_NODES * DK, kTb, N_NODES, (long long)DK * N_NODES,
          N_NODES, N_NODES, DK, epi);
    }
    softmax_rows_bf<<<dim3(NH * N_NODES), blk, 0, stream>>>(attn, attnbf, N_NODES);

    { // o[m, h*dk + k] = attn_bf[h] @ V[h]  -> bf16 scatter
      EpiOBf epi{obuf, NH * DK, DK};
      gemm_wmma<<<dim3(DK / 128, N_NODES / 64, NH), blk, 0, stream>>>(
          attnbf, N_NODES, (long long)N_NODES * N_NODES, vbuf, DK, (long long)N_NODES * DK,
          N_NODES, DK, N_NODES, epi);
    }
    { // h = o @ Wo[l] + bo + h   (residual fused, in-place safe)
      EpiBiasRes epi{hbuf, ND, bo + (size_t)l * ND, hbuf, ND};
      gemm_wmma<<<dim3(ND / 128, N_NODES / 64, 1), blk, 0, stream>>>(
          obuf, NH * DK, 0LL, Wo + (size_t)l * NH * ND * ND, ND, 0LL,
          N_NODES, ND, NH * DK, epi);
    }

    ln_rows<<<dim3(N_NODES), blk, 0, stream>>>(hbuf, xnb, ln2_w + (size_t)l * ND, ln2_b + (size_t)l * ND, ND);

    { // ff = gelu(x2 @ Wff1 + bff1) -> bf16
      EpiGeluBf epi{ffb, FF, bff1 + (size_t)l * FF};
      gemm_wmma<<<dim3(FF / 128, N_NODES / 64, 1), blk, 0, stream>>>(
          xnb, ND, 0LL, Wff1 + (size_t)l * ND * FF, FF, 0LL, N_NODES, FF, ND, epi);
    }
    { // h = ff @ Wff2 + bff2 + h
      EpiBiasRes epi{hbuf, ND, bff2 + (size_t)l * ND, hbuf, ND};
      gemm_wmma<<<dim3(ND / 128, N_NODES / 64, 1), blk, 0, stream>>>(
          ffb, FF, 0LL, Wff2 + (size_t)l * FF * ND, ND, 0LL, N_NODES, ND, FF, epi);
    }
  }

  { // out = h @ Wout + bout
    EpiBias epi{(float*)d_out, 0LL, OUT_D, bout, 0};
    gemm_wmma<<<dim3(OUT_D / 128, N_NODES / 64, 1), blk, 0, stream>>>(
        hbuf, ND, 0LL, Wout, OUT_D, 0LL, N_NODES, OUT_D, ND, epi);
  }
}
